// CepstrumToImpulseResponse_47107201302885
// MI455X (gfx1250) — matrix operand order repaired and checked by reading, not compiled, not run on hardware
//
#include <hip/hip_runtime.h>
#include <math.h>

// CDNA5 / gfx1250 — wave32, WMMA (not MFMA).
typedef __attribute__((ext_vector_type(16))) __bf16 v16bf;
typedef __attribute__((ext_vector_type(8)))  float  v8f;

#define CEP_M           99            // cepstral order M
#define NCOEF           (CEP_M + 1)   // 100 input coeffs per batch
#define IR_N            512           // output IR length
#define HIST            112           // history window (7 chunks of 16, >= M)
#define WAVES_PER_BLOCK 8

// per-wave LDS partition (floats); stride kept a multiple of 4 floats (16B)
#define WF_OFF   0                    // wf[0..111]  : w[m] = m*c[m], zero-padded
#define HB_OFF   112                  // hb[0..623]  : hb[i] = h[i-112]; hb[0..111]=0
#define WAVE_LDS 752

// Per-wave LDS ordering: DS ops of one wave complete in order; the explicit
// s_wait_dscnt + scheduling barrier makes lane0's stores visible to the other
// lanes' loads without coupling the 8 independent waves of the workgroup.
static __device__ __forceinline__ void wave_sync_lds() {
  __builtin_amdgcn_wave_barrier();
  asm volatile("s_wait_dscnt 0x0" ::: "memory");
  __builtin_amdgcn_wave_barrier();
}

// broadcast a float from a compile-time lane: v_readlane_b32 (no LDS traffic)
static __device__ __forceinline__ float bcast_lane(float v, int srcLane) {
  union { float f; int i; } u;
  u.f = v;
  u.i = __builtin_amdgcn_readlane(u.i, srcLane);
  return u.f;
}

__global__ __launch_bounds__(256) void
CepstrumToImpulseResponse_gfx1250(const float* __restrict__ c,
                                  float* __restrict__ out,
                                  int batch) {
  __shared__ __align__(16) float lds[WAVES_PER_BLOCK][WAVE_LDS];

  const int lane = threadIdx.x & 31;
  const int wid  = threadIdx.x >> 5;
  int b = blockIdx.x * WAVES_PER_BLOCK + wid;
  if (b >= batch) b = batch - 1;     // tail waves duplicate last batch (benign)

  const float* cb = c + (size_t)b * NCOEF;
  float* ob = out + (size_t)b * IR_N;
  float* wf = &lds[wid][WF_OFF];
  float* hb = &lds[wid][HB_OFF];

  __builtin_prefetch(cb, 0, 3);      // global_prefetch_b8

  // ---- init: weighted cepstrum + zeroed h buffer (112-zero prologue) ----
  for (int m = lane; m < HIST; m += 32)
    wf[m] = (m >= 1 && m <= CEP_M) ? (float)m * cb[m] : 0.0f;
  {
    float4 z = make_float4(0.f, 0.f, 0.f, 0.f);
    float4* hz = (float4*)hb;
    for (int k = lane; k < (HIST + IR_N) / 4; k += 32) hz[k] = z;   // b128 stores
  }
  wave_sync_lds();

  const int hi  = lane >> 4;         // 0: K-low half, 1: K-high half
  const int row = lane & 15;

  // ---- A (bf16 16x128 Toeplitz of w), built ONCE, branchless ----
  // A[row][j] = w[112+row-j] iff j<112 and 1<=112+row-j<=99, else 0.
  union VA { v16bf v; __bf16 e[16]; };
  VA a[4];
#pragma unroll
  for (int s = 0; s < 4; ++s) {
#pragma unroll
    for (int t = 0; t < 16; ++t) {
      int kl = 16 * hi + t;                       // K within the 32-slab
      int j  = 32 * s + kl;                       // global K index
      int m  = HIST + row - j;                    // tap index
      int mc = m < 0 ? 0 : (m > 111 ? 111 : m);   // clamp for a safe LDS read
      float v = wf[mc];                           // wf[0]=0, wf[100..111]=0
      v = (j < HIST && m >= 1 && m <= CEP_M) ? v : 0.0f;   // v_cndmask
      a[s].e[t] = (__bf16)v;
    }
  }

  // intra-block taps m=1..15 live in registers (all lanes)
  float wfr[16];
#pragma unroll
  for (int m = 0; m < 16; ++m) wfr[m] = wf[m];

  // ---- block 0 warm-up: registers only ----
  float hloc[16];
  hloc[0] = expf(cb[0]);             // h[0] = exp(c0)
#pragma unroll
  for (int n = 1; n < 16; ++n) {
    float sum = 0.0f;
#pragma unroll
    for (int m = 1; m <= n; ++m) sum += wfr[m] * hloc[n - m];
    hloc[n] = sum * __builtin_amdgcn_rcpf((float)n);
  }
  if (lane == 0) {
    float4* dst = (float4*)(hb + HIST);
    dst[0] = make_float4(hloc[0],  hloc[1],  hloc[2],  hloc[3]);
    dst[1] = make_float4(hloc[4],  hloc[5],  hloc[6],  hloc[7]);
    dst[2] = make_float4(hloc[8],  hloc[9],  hloc[10], hloc[11]);
    dst[3] = make_float4(hloc[12], hloc[13], hloc[14], hloc[15]);
  }

  // ---- blocks 1..31: WMMA history matvec + register triangular solve ----
  for (int p = 1; p < IR_N / 16; ++p) {
    const int t0 = 16 * p;
    wave_sync_lds();                 // previous block's hb stores visible

    // B: ALL lanes load the SAME history window (broadcast) -> every column
    // of D equals y; no masking needed.  16 consecutive f32 per slab per lane.
    union VB { v16bf v; __bf16 e[16]; };
    v8f acc = {};
#pragma unroll
    for (int s = 0; s < 4; ++s) {
      const float4* src = (const float4*)(hb + t0 + 32 * s + 16 * hi);
      float4 f0 = src[0], f1 = src[1], f2 = src[2], f3 = src[3];  // ds_load_b128
      VB bs;
      bs.e[0]  = (__bf16)f0.x; bs.e[1]  = (__bf16)f0.y;
      bs.e[2]  = (__bf16)f0.z; bs.e[3]  = (__bf16)f0.w;
      bs.e[4]  = (__bf16)f1.x; bs.e[5]  = (__bf16)f1.y;
      bs.e[6]  = (__bf16)f1.z; bs.e[7]  = (__bf16)f1.w;
      bs.e[8]  = (__bf16)f2.x; bs.e[9]  = (__bf16)f2.y;
      bs.e[10] = (__bf16)f2.z; bs.e[11] = (__bf16)f2.w;
      bs.e[12] = (__bf16)f3.x; bs.e[13] = (__bf16)f3.y;
      bs.e[14] = (__bf16)f3.z; bs.e[15] = (__bf16)f3.w;
      acc = __builtin_amdgcn_wmma_f32_16x16x32_bf16(
                /*neg_a=*/false, a[s].v, /*neg_b=*/false, bs.v,
                /*c_mod=*/(short)0, acc, /*reuse_a=*/false, /*reuse_b=*/false);
    }

    // gather y via v_readlane_b32: lane0 holds rows 0..7, lane16 rows 8..15
    union VD { v8f v; float e[8]; } d;
    d.v = acc;
    float y[16];
#pragma unroll
    for (int r = 0; r < 8; ++r) {
      y[r]     = bcast_lane(d.e[r], 0);
      y[r + 8] = bcast_lane(d.e[r], 16);
    }

    // 16-step strictly-lower-triangular solve, registers only
    // (compiler sinks this under the lane-0 liveness mask)
#pragma unroll
    for (int i = 0; i < 16; ++i) {
      float sum = y[i];
#pragma unroll
      for (int m = 1; m <= i; ++m) sum += wfr[m] * hloc[i - m];
      hloc[i] = sum * __builtin_amdgcn_rcpf((float)(t0 + i));
    }

    if (lane == 0) {
      float4* dst = (float4*)(hb + HIST + t0);
      dst[0] = make_float4(hloc[0],  hloc[1],  hloc[2],  hloc[3]);
      dst[1] = make_float4(hloc[4],  hloc[5],  hloc[6],  hloc[7]);
      dst[2] = make_float4(hloc[8],  hloc[9],  hloc[10], hloc[11]);
      dst[3] = make_float4(hloc[12], hloc[13], hloc[14], hloc[15]);
    }
  }
  wave_sync_lds();

  // ---- vectorized writeback: 512 f32 per batch as float4 (b128 stores) ----
  const float4* hv = (const float4*)(hb + HIST);
  float4* ov = (float4*)ob;
  for (int k = lane; k < IR_N / 4; k += 32) ov[k] = hv[k];
}

extern "C" void kernel_launch(void* const* d_in, const int* in_sizes, int n_in,
                              void* d_out, int out_size, void* d_ws, size_t ws_size,
                              hipStream_t stream) {
  const float* c = (const float*)d_in[0];
  float* out = (float*)d_out;
  const int batch = in_sizes[0] / NCOEF;          // 65536
  const int nblocks = (batch + WAVES_PER_BLOCK - 1) / WAVES_PER_BLOCK;
  hipLaunchKernelGGL(CepstrumToImpulseResponse_gfx1250,
                     dim3(nblocks), dim3(32 * WAVES_PER_BLOCK), 0, stream,
                     c, out, batch);
}
